// Gtu_52355651338764
// MI455X (gfx1250) — compile-verified, matrix operand order checked by reading, not run on hardware
//
#include <hip/hip_runtime.h>
#include <hip/hip_bf16.h>
#include <math.h>

typedef __attribute__((ext_vector_type(16))) __bf16 v16bf;
typedef __attribute__((ext_vector_type(8)))  float  v8f;

#define B_   8
#define HH_  128
#define WW_  128
#define E_   192
#define D1_  576
#define NH_  8
#define HD_  72
#define TOK_ 131072   /* B*H*W */
#define RPE_ 512

#define LSTR 24       /* per-lane fragment stride in halfs: 48B, 16B aligned */

__device__ __forceinline__ float  bf2f(__bf16 v) { return (float)v; }
__device__ __forceinline__ __bf16 f2bf(float v)  { return (__bf16)v; }

// Load a 32B WMMA fragment (16 bf16) from LDS with two b128 reads.
__device__ __forceinline__ v16bf ld_frag(const __bf16* p) {
    union { uint4 u[2]; v16bf v; } t;
    t.u[0] = *(const uint4*)p;
    t.u[1] = *(const uint4*)(p + 8);
    return t.v;
}

// Convert 8 contiguous f32 to 8 packed bf16 (one uint4).
__device__ __forceinline__ uint4 pack8f(const float* s) {
    union { __bf16 h[8]; uint4 u; } t;
#pragma unroll
    for (int j = 0; j < 8; ++j) t.h[j] = f2bf(s[j]);
    return t.u;
}

enum { ACT_NONE = 0, ACT_SILU = 1 };
enum { ST_F32 = 0, ST_BF16 = 1, ST_V = 2, ST_RPE = 3, ST_FINAL = 4 };

// ---------------------------------------------------------------------------
// bf16 WMMA GEMM:  C[M,N] = act(A[M,K] (bf16) * W[N,K]^T (f32) + bias)
// FUSE: A_eff[m][k] = A[m*K+k] * A2[k*M+m]   (u * tno, tno channel-major)
// Block: 256 thr (8 waves), tile 256(M) x 64(N), BK=32; wave computes 32x64.
// All LDS tiles stored in WMMA fragment layout: (tileIdx*32 + lane)*LSTR + j,
// where lane = (idx16) + hi*16, hi = (k>>3)&1, j = ((k>>4)<<3) + (k&7).
// ---------------------------------------------------------------------------
template<int ACT, int STORE, bool FUSE>
__global__ __launch_bounds__(256) void gemm_wmma(
    const __bf16* __restrict__ A, const __bf16* __restrict__ A2,
    const float* __restrict__ Wt, const float* __restrict__ Bias,
    float* __restrict__ outF, __bf16* __restrict__ outB, __bf16* __restrict__ outB2,
    const float* __restrict__ shortcut,
    int M, int N, int K)
{
    __shared__ __align__(16) __bf16 AsF[16 * 32 * LSTR];               // 24KB
    __shared__ __align__(16) __bf16 AtF[FUSE ? 16 * 32 * LSTR : 16];  // 24KB if FUSE
    __shared__ __align__(16) __bf16 WsF[4 * 32 * LSTR];               // 6KB

    const int tid  = threadIdx.x;
    const int lane = tid & 31;
    const int wave = tid >> 5;
    const int n0 = blockIdx.x * 64;
    const int m0 = blockIdx.y * 256;

    v8f acc[2][4];
#pragma unroll
    for (int s = 0; s < 2; ++s)
#pragma unroll
        for (int t = 0; t < 4; ++t)
#pragma unroll
            for (int e = 0; e < 8; ++e) acc[s][t][e] = 0.0f;

    const int nk = K >> 5;
    for (int kt = 0; kt < nk; ++kt) {
        const int k0 = kt << 5;
        // ---- stage A row (tid = row, 32 halfs = 4 x uint4), frag layout ----
        {
            const __bf16* src = A + (size_t)(m0 + tid) * K + k0;
            uint4 p0 = *(const uint4*)(src);
            uint4 p1 = *(const uint4*)(src + 8);
            uint4 p2 = *(const uint4*)(src + 16);
            uint4 p3 = *(const uint4*)(src + 24);
            const int lb = (tid >> 4) * 32 + (tid & 15);
            *(uint4*)&AsF[(lb     ) * LSTR    ] = p0;   // k 0..7   hi=0 j=0
            *(uint4*)&AsF[(lb + 16) * LSTR    ] = p1;   // k 8..15  hi=1 j=0
            *(uint4*)&AsF[(lb     ) * LSTR + 8] = p2;   // k 16..23 hi=0 j=8
            *(uint4*)&AsF[(lb + 16) * LSTR + 8] = p3;   // k 24..31 hi=1 j=8
        }
        // ---- stage A2 (channel-major, needs transpose): 4k x 8m per thread ----
        if (FUSE) {
            const int mb  = (tid >> 3) * 8;   // m base (0..248)
            const int kb4 = (tid & 7) * 4;    // k base (0..28)
            union { uint4 u; __bf16 h[8]; } rv[4];
#pragma unroll
            for (int r = 0; r < 4; ++r)
                rv[r].u = *(const uint4*)(A2 + (size_t)(k0 + kb4 + r) * M + (m0 + mb));
            const int hi = (kb4 >> 3) & 1;
            const int jj = ((kb4 >> 4) << 3) + (kb4 & 7);
#pragma unroll
            for (int m = 0; m < 8; ++m) {
                union { __bf16 h[4]; uint2 u; } o;
#pragma unroll
                for (int r = 0; r < 4; ++r) o.h[r] = rv[r].h[m];
                const int li = ((mb + m) >> 4) * 32 + hi * 16 + ((mb + m) & 15);
                *(uint2*)&AtF[li * LSTR + jj] = o.u;
            }
        }
        // ---- stage W tile (f32->bf16, 8 contiguous k per thread) ----
        {
            const int ni = tid >> 2;
            const int kc = (tid & 3) << 3;
            uint4 w8 = pack8f(Wt + (size_t)(n0 + ni) * K + (k0 + kc));
            const int hi = (kc >> 3) & 1;
            const int jb = (kc >> 4) << 3;
            const int li = (ni >> 4) * 32 + hi * 16 + (ni & 15);
            *(uint4*)&WsF[li * LSTR + jb] = w8;
        }
        __syncthreads();

        // ---- compute ----
        v16bf bfr[4];
#pragma unroll
        for (int t = 0; t < 4; ++t)
            bfr[t] = ld_frag(&WsF[(t * 32 + lane) * LSTR]);
#pragma unroll
        for (int s = 0; s < 2; ++s) {
            const int strip = (wave << 1) + s;
            v16bf af = ld_frag(&AsF[(strip * 32 + lane) * LSTR]);
            if (FUSE) {
                v16bf tf = ld_frag(&AtF[(strip * 32 + lane) * LSTR]);
                af = af * tf;   // elementwise bf16 multiply (v_pk_mul_bf16 capable)
            }
#pragma unroll
            for (int t = 0; t < 4; ++t)
                acc[s][t] = __builtin_amdgcn_wmma_f32_16x16x32_bf16(
                    false, af, false, bfr[t], (short)0, acc[s][t], false, false);
        }
        __syncthreads();
    }

    // ---- epilogue ----
    const int hi8 = (lane >> 4) << 3;
    const int lm = lane & 15;
#pragma unroll
    for (int s = 0; s < 2; ++s) {
#pragma unroll
        for (int t = 0; t < 4; ++t) {
#pragma unroll
            for (int e = 0; e < 8; ++e) {
                const int row = m0 + ((wave << 1) + s) * 16 + e + hi8;
                const int col = n0 + (t << 4) + lm;
                float val = acc[s][t][e] + Bias[col];
                if (ACT == ACT_SILU) val = val / (1.0f + expf(-val));
                if (STORE == ST_F32) {
                    outF[(size_t)row * N + col] = val;
                } else if (STORE == ST_BF16) {
                    outB[(size_t)row * N + col] = f2bf(val);
                } else if (STORE == ST_V) {
                    const int bb = row >> 14;
                    const int ii = (row >> 7) & 127;
                    const int ww = row & 127;
                    const __bf16 bv = f2bf(val);
                    outB [(size_t)col * TOK_ + ii * 1024 + bb * 128 + ww] = bv;
                    outB2[(size_t)col * TOK_ + ww * 1024 + bb * 128 + ii] = bv;
                } else if (STORE == ST_RPE) {
                    const int head = col / HD_;
                    const int dd = col - head * HD_;
                    outF[(size_t)(head * 256 + row) * HD_ + dd] = val;
                } else { // ST_FINAL
                    outF[(size_t)row * N + col] = val + shortcut[(size_t)row * N + col];
                }
            }
        }
    }
}

// ---------------------------------------------------------------------------
// Toeplitz mixing per (hd, b):  C1 = T2 @ vH_slice,  C2 = T1 @ vW_slice,
//   out[b,i,w,hd] = C1[i][w] + C2[w][i]  -> tnoT[hd*TOK + token] (bf16)
// Toeplitz A-fragments read from a REVERSED bf16 coefficient table so the
// 16 fragment elements are ascending-contiguous in LDS (wide loads, no swaps):
//   coefR[r] = coef[(256 - r) & 255]  =>  T[m][k] = coefR[k - m + 256].
// ---------------------------------------------------------------------------
__global__ __launch_bounds__(256) void toeplitz_wmma(
    const float* __restrict__ a1, const float* __restrict__ a2,
    const __bf16* __restrict__ vH, const __bf16* __restrict__ vW,
    __bf16* __restrict__ tnoT)
{
    __shared__ __align__(16) __bf16 coefR1[512];
    __shared__ __align__(16) __bf16 coefR2[512];
    __shared__ __align__(16) char smem[2 * 2 * 8 * 32 * LSTR * 2];  // 49152B
    __bf16* B1F = (__bf16*)smem;                     // [ks2][t][lane][LSTR]
    __bf16* B2F = B1F + 2 * 8 * 32 * LSTR;
    __bf16* C2s = (__bf16*)smem;                     // reuse: 128 x 129 halfs

    const int tid  = threadIdx.x;
    const int lane = tid & 31;
    const int wave = tid >> 5;
    const int hd = blockIdx.x;
    const int b  = blockIdx.y;
    const int head = hd / HD_;
    const int d = hd - head * HD_;

    {
        const __bf16 c1 = f2bf(a1[(size_t)(head * 256 + tid) * HD_ + d]);
        const __bf16 c2 = f2bf(a2[(size_t)(head * 256 + tid) * HD_ + d]);
        const int r0 = (256 - tid) & 255;
        coefR1[r0] = c1; coefR1[r0 + 256] = c1;
        coefR2[r0] = c2; coefR2[r0 + 256] = c2;
    }

    v8f acc1[8], acc2[8];
#pragma unroll
    for (int t = 0; t < 8; ++t)
#pragma unroll
        for (int e = 0; e < 8; ++e) { acc1[t][e] = 0.0f; acc2[t][e] = 0.0f; }

    const int mrow = (wave << 4) + (lane & 15);
    const int g = (lane >> 4) << 3;

    for (int half = 0; half < 2; ++half) {
        __syncthreads();
        {   // stage 64 K-rows of vH/vW slices into fragment layout
            const int jp = tid >> 3;          // 0..31 -> k-row pair
            const int nc = (tid & 7) << 4;    // n base, 16 wide
            const int jl = jp << 1;           // local k row (even)
            const int jg = (half << 6) + jl;
            const int kl = jl & 31;
            const int ks2 = jl >> 5;
            const int hi = (kl >> 3) & 1;
            const int jj = ((kl >> 4) << 3) + (kl & 7);
            const size_t gofs = (size_t)hd * TOK_ + (size_t)jg * 1024 + b * 128 + nc;
            union { uint4 u; __bf16 h[8]; } q[2][2][2]; // [tile][krow][nhalf]
            q[0][0][0].u = *(const uint4*)(vH + gofs);
            q[0][0][1].u = *(const uint4*)(vH + gofs + 8);
            q[0][1][0].u = *(const uint4*)(vH + gofs + 1024);
            q[0][1][1].u = *(const uint4*)(vH + gofs + 1032);
            q[1][0][0].u = *(const uint4*)(vW + gofs);
            q[1][0][1].u = *(const uint4*)(vW + gofs + 8);
            q[1][1][0].u = *(const uint4*)(vW + gofs + 1024);
            q[1][1][1].u = *(const uint4*)(vW + gofs + 1032);
#pragma unroll
            for (int nn = 0; nn < 16; ++nn) {
                const int n = nc + nn;
                const int li = ((ks2 * 8 + (n >> 4)) * 32 + hi * 16 + (n & 15)) * LSTR + jj;
                union { __bf16 h[2]; unsigned u; } pk;
                pk.h[0] = q[0][0][nn >> 3].h[nn & 7];
                pk.h[1] = q[0][1][nn >> 3].h[nn & 7];
                *(unsigned*)&B1F[li] = pk.u;
                pk.h[0] = q[1][0][nn >> 3].h[nn & 7];
                pk.h[1] = q[1][1][nn >> 3].h[nn & 7];
                *(unsigned*)&B2F[li] = pk.u;
            }
        }
        __syncthreads();
#pragma unroll
        for (int ks2 = 0; ks2 < 2; ++ks2) {
            const int kg = (half << 6) + (ks2 << 5);
            const int base = kg - mrow + 256;     // ascending index: + kb
            v16bf a1f, a2f;
#pragma unroll
            for (int r = 0; r < 8; ++r) {
                const int kb = ((r & 4) << 2) + g + ((r & 3) << 1);
                a2f[2 * r]     = coefR2[base + kb];
                a2f[2 * r + 1] = coefR2[base + kb + 1];
                a1f[2 * r]     = coefR1[base + kb];
                a1f[2 * r + 1] = coefR1[base + kb + 1];
            }
#pragma unroll
            for (int t = 0; t < 8; ++t) {
                v16bf b1f = ld_frag(&B1F[((ks2 * 8 + t) * 32 + lane) * LSTR]);
                v16bf b2f = ld_frag(&B2F[((ks2 * 8 + t) * 32 + lane) * LSTR]);
                acc1[t] = __builtin_amdgcn_wmma_f32_16x16x32_bf16(
                    false, a2f, false, b1f, (short)0, acc1[t], false, false);
                acc2[t] = __builtin_amdgcn_wmma_f32_16x16x32_bf16(
                    false, a1f, false, b2f, (short)0, acc2[t], false, false);
            }
        }
    }
    __syncthreads();
    const int hi8 = (lane >> 4) << 3;
    const int lm = lane & 15;
#pragma unroll
    for (int t = 0; t < 8; ++t)
#pragma unroll
        for (int e = 0; e < 8; ++e) {
            const int r_ = (wave << 4) + e + hi8;
            const int c_ = (t << 4) + lm;
            C2s[r_ * 129 + c_] = f2bf(acc2[t][e]);
        }
    __syncthreads();
    const size_t base = (size_t)hd * TOK_ + (size_t)b * (HH_ * WW_);
#pragma unroll
    for (int t = 0; t < 8; ++t)
#pragma unroll
        for (int e = 0; e < 8; ++e) {
            const int i = (wave << 4) + e + hi8;
            const int w = (t << 4) + lm;
            const float val = acc1[t][e] + bf2f(C2s[w * 129 + i]);
            tnoT[base + i * 128 + w] = f2bf(val);
        }
}

// ---------------------------------------------------------------------------
// SimpleRMSNorm of x (dim 192), wave per row, -> bf16
// ---------------------------------------------------------------------------
__global__ __launch_bounds__(256) void srms_x_kernel(
    const float* __restrict__ x, __bf16* __restrict__ xn)
{
    const int lane = threadIdx.x & 31;
    const int wave = threadIdx.x >> 5;
    const size_t row = (size_t)blockIdx.x * 8 + wave;
    const float* p = x + row * E_;
    float v[6];
    float s = 0.f;
#pragma unroll
    for (int j = 0; j < 6; ++j) { v[j] = p[lane + 32 * j]; s += v[j] * v[j]; }
#pragma unroll
    for (int o = 16; o > 0; o >>= 1) s += __shfl_xor(s, o, 32);
    const float scale = 1.0f / (sqrtf(s) * 0.07216878364870322f + 1e-8f);
    __bf16* q = xn + row * E_;
#pragma unroll
    for (int j = 0; j < 6; ++j) q[lane + 32 * j] = f2bf(v[j] * scale);
}

// relu(srms(h)) for RPE hidden (dim 512), wave per row, -> bf16
__global__ __launch_bounds__(256) void srms_relu_kernel(
    const float* __restrict__ h, __bf16* __restrict__ o)
{
    const int lane = threadIdx.x & 31;
    const int wave = threadIdx.x >> 5;
    const size_t row = (size_t)blockIdx.x * 8 + wave;
    const float* p = h + row * RPE_;
    float v[16];
    float s = 0.f;
#pragma unroll
    for (int j = 0; j < 16; ++j) { v[j] = p[lane + 32 * j]; s += v[j] * v[j]; }
#pragma unroll
    for (int oo = 16; oo > 0; oo >>= 1) s += __shfl_xor(s, oo, 32);
    const float scale = 1.0f / (sqrtf(s) * 0.04419417382415922f + 1e-8f);
    __bf16* q = o + row * RPE_;
#pragma unroll
    for (int j = 0; j < 16; ++j) {
        const float t = v[j] * scale;
        q[lane + 32 * j] = f2bf(t > 0.f ? t : 0.f);
    }
}

// h0[row][c] = pos(row) * pos_w[c] + pos_b[c];  pos layout [0,1..127,0,-127..-1]
__global__ __launch_bounds__(512) void rpe_pos_kernel(
    const float* __restrict__ pos_w, const float* __restrict__ pos_b,
    float* __restrict__ h0)
{
    const int row = blockIdx.x;
    const int c = threadIdx.x;
    float pos;
    if (row == 0 || row == 128) pos = 0.f;
    else if (row < 128)         pos = (float)row;
    else                        pos = (float)(row - 256);
    h0[(size_t)row * RPE_ + c] = pos * pos_w[c] + pos_b[c];
}

// ---------------------------------------------------------------------------
extern "C" void kernel_launch(void* const* d_in, const int* in_sizes, int n_in,
                              void* d_out, int out_size, void* d_ws, size_t ws_size,
                              hipStream_t stream)
{
    const float* x   = (const float*)d_in[0];
    const float* u_w = (const float*)d_in[1];
    const float* u_b = (const float*)d_in[2];
    const float* v_w = (const float*)d_in[3];
    const float* v_b = (const float*)d_in[4];
    const float* o_w = (const float*)d_in[5];
    const float* o_b = (const float*)d_in[6];
    const float* rpw[2] = { (const float*)d_in[7],  (const float*)d_in[13] };
    const float* rpb[2] = { (const float*)d_in[8],  (const float*)d_in[14] };
    const float* rlw[2] = { (const float*)d_in[9],  (const float*)d_in[15] };
    const float* rlb[2] = { (const float*)d_in[10], (const float*)d_in[16] };
    const float* row_w[2] = { (const float*)d_in[11], (const float*)d_in[17] };
    const float* row_b[2] = { (const float*)d_in[12], (const float*)d_in[18] };
    float* out = (float*)d_out;

    char* ws = (char*)d_ws;
    size_t off = 0;
    auto alloc = [&](size_t bytes) -> char* {
        char* p = ws + off;
        off += (bytes + 255) & ~(size_t)255;
        return p;
    };
    __bf16* xn   = (__bf16*)alloc((size_t)TOK_ * E_ * 2);
    __bf16* ubuf = (__bf16*)alloc((size_t)TOK_ * D1_ * 2);
    __bf16* vH   = (__bf16*)alloc((size_t)D1_ * TOK_ * 2);
    __bf16* vW   = (__bf16*)alloc((size_t)D1_ * TOK_ * 2);
    __bf16* tno  = (__bf16*)alloc((size_t)D1_ * TOK_ * 2);
    float*  h0   = (float*)alloc((size_t)256 * RPE_ * 4);
    __bf16* hbf  = (__bf16*)alloc((size_t)256 * RPE_ * 2);
    float*  abuf0 = (float*)alloc((size_t)NH_ * 256 * HD_ * 4);
    float*  abuf1 = (float*)alloc((size_t)NH_ * 256 * HD_ * 4);
    float*  abuf[2] = { abuf0, abuf1 };

    // 1) pre-norm
    srms_x_kernel<<<TOK_ / 8, 256, 0, stream>>>(x, xn);

    // 2) u = silu(xn @ u_w^T + u_b)  (token-major bf16)
    gemm_wmma<ACT_SILU, ST_BF16, false><<<dim3(D1_ / 64, TOK_ / 256), 256, 0, stream>>>(
        xn, nullptr, u_w, u_b, nullptr, ubuf, nullptr, nullptr, TOK_, D1_, E_);

    // 3) v = silu(xn @ v_w^T + v_b)  scattered to both Toeplitz layouts
    gemm_wmma<ACT_SILU, ST_V, false><<<dim3(D1_ / 64, TOK_ / 256), 256, 0, stream>>>(
        xn, nullptr, v_w, v_b, nullptr, vH, vW, nullptr, TOK_, D1_, E_);

    // 4) RPE nets (net 0 = rpe1 / W-axis, net 1 = rpe2 / H-axis)
    for (int net = 0; net < 2; ++net) {
        rpe_pos_kernel<<<256, 512, 0, stream>>>(rpw[net], rpb[net], h0);
        for (int l = 0; l < 3; ++l) {
            srms_relu_kernel<<<32, 256, 0, stream>>>(h0, hbf);
            gemm_wmma<ACT_NONE, ST_F32, false><<<dim3(RPE_ / 64, 1), 256, 0, stream>>>(
                hbf, nullptr, rlw[net] + (size_t)l * RPE_ * RPE_, rlb[net] + (size_t)l * RPE_,
                h0, nullptr, nullptr, nullptr, 256, RPE_, RPE_);
        }
        srms_relu_kernel<<<32, 256, 0, stream>>>(h0, hbf);
        gemm_wmma<ACT_NONE, ST_RPE, false><<<dim3(D1_ / 64, 1), 256, 0, stream>>>(
            hbf, nullptr, row_w[net], row_b[net],
            abuf[net], nullptr, nullptr, nullptr, 256, D1_, RPE_);
    }

    // 5) Toeplitz mixing along H (a2) and W (a1), summed, channel-major bf16
    toeplitz_wmma<<<dim3(D1_, B_), 256, 0, stream>>>(abuf[0], abuf[1], vH, vW, tno);

    // 6) out = (u * tno) @ o_w^T + o_b + x
    gemm_wmma<ACT_NONE, ST_FINAL, true><<<dim3(E_ / 64, TOK_ / 256), 256, 0, stream>>>(
        ubuf, tno, o_w, o_b, out, nullptr, nullptr, x, TOK_, E_, D1_);
}
